// MultiHeadAttention_50577534878009
// MI455X (gfx1250) — matrix operand order repaired and checked by reading, not compiled, not run on hardware
//
#include <hip/hip_runtime.h>
#include <hip/hip_bf16.h>

typedef _Float16 half_t;
typedef __attribute__((ext_vector_type(16))) _Float16 v16h;
typedef __attribute__((ext_vector_type(8)))  _Float16 v8h;
typedef __attribute__((ext_vector_type(4)))  _Float16 v4h;
typedef __attribute__((ext_vector_type(8)))  float    v8f;

#define WMMA_F16(a, b, c) \
    __builtin_amdgcn_wmma_f32_16x16x32_f16(false, (a), false, (b), (short)0, (c), false, false)

// ---------------------------------------------------------------------------
// CDNA5 async global->LDS copy (16 bytes per lane), tracked by ASYNCcnt.
// ---------------------------------------------------------------------------
__device__ __forceinline__ void async_copy16(void* lptr, const void* gptr) {
    unsigned lds = (unsigned)(unsigned long long)lptr;   // low 32 bits = LDS offset
    asm volatile("global_load_async_to_lds_b128 %0, %1, off"
                 :: "v"(lds), "v"(gptr) : "memory");
}
__device__ __forceinline__ void async_wait0() {
    asm volatile("s_wait_asynccnt 0x0" ::: "memory");
}

// ---------------------------------------------------------------------------
// Fragment loaders (CDNA5 16x16x32 f16 layouts, ISA 7.12.2)
// A fragment: lane r (0..15) half h: elems 0..7 = A[r][8h..8h+7],
//                                    elems 8..15 = A[r][16+8h..16+8h+7]
// B fragment (from [N][K] tile): lane r half h: elems 0..7 = Bt[r][8h..8h+7],
//                                             elems 8..15 = Bt[r][16+8h..16+8h+7]
// ---------------------------------------------------------------------------
__device__ __forceinline__ v16h ld_afrag(const half_t* base, int ld, int lane) {
    int r = lane & 15, h = lane >> 4;
    const half_t* p = base + r * ld + 8 * h;
    v8h lo = *(const v8h*)(p);
    v8h hi = *(const v8h*)(p + 16);
    v16h a;
#pragma unroll
    for (int i = 0; i < 8; ++i) { a[i] = lo[i]; a[8 + i] = hi[i]; }
    return a;
}

__device__ __forceinline__ v16h ld_bfrag(const half_t* base, int ld, int lane) {
    int r = lane & 15, h = lane >> 4;
    const half_t* p = base + r * ld + 8 * h;
    v8h lo = *(const v8h*)(p);
    v8h hi = *(const v8h*)(p + 16);
    v16h b;
#pragma unroll
    for (int i = 0; i < 8; ++i) { b[i] = lo[i]; b[8 + i] = hi[i]; }
    return b;
}

// ---------------------------------------------------------------------------
// One-time f32 -> f16 conversion (elementwise, vectorized).
// ---------------------------------------------------------------------------
__global__ __launch_bounds__(256) void cvt_f16(const float* __restrict__ in,
                                               half_t* __restrict__ out) {
    size_t i = ((size_t)blockIdx.x * 256 + threadIdx.x) * 4;
    float4 x = *(const float4*)(in + i);
    v4h p = { (half_t)x.x, (half_t)x.y, (half_t)x.z, (half_t)x.w };
    *(v4h*)(out + i) = p;
}

// ---------------------------------------------------------------------------
// One-time weight convert + transpose: W[1024 k][1024 n] f32 -> Wt[n][k] f16.
// 32x32 LDS tile transpose, 256 threads.
// ---------------------------------------------------------------------------
__global__ __launch_bounds__(256) void tcvt_f16(const float* __restrict__ in,
                                                half_t* __restrict__ out) {
    __shared__ float t[32][33];
    const int tx = threadIdx.x & 31, ty = threadIdx.x >> 5;   // ty 0..7
    const int k0 = blockIdx.y * 32, n0 = blockIdx.x * 32;
#pragma unroll
    for (int i = 0; i < 4; ++i)
        t[ty + i * 8][tx] = in[(size_t)(k0 + ty + i * 8) * 1024 + n0 + tx];
    __syncthreads();
#pragma unroll
    for (int i = 0; i < 4; ++i)
        out[(size_t)(n0 + ty + i * 8) * 1024 + k0 + tx] = (half_t)t[tx][ty + i * 8];
}

// ---------------------------------------------------------------------------
// WMMA GEMM, both operands f16, B pre-transposed to [N][K].
// Inner loop = async global->LDS (double-buffered) + 8 WMMA per wave.
//   OMODE = 0: output f16 scattered to [b, h, s, d]  (B=8,H=16,S=1024,D=64)
//   OMODE = 1: output f32 row-major [M, N]
// Block 256 threads = 8 waves; block tile 128x128; wave tile 64x32 (4x2 WMMA)
// ---------------------------------------------------------------------------
template <int OMODE>
__global__ __launch_bounds__(256) void gemm_wmma(const half_t* __restrict__ A,
                                                 const half_t* __restrict__ Bt,
                                                 void* __restrict__ Op,
                                                 int M, int N, int K) {
    __shared__ half_t As[2][128][40];
    __shared__ half_t Bs[2][128][40];

    const int tid  = threadIdx.x;
    const int w    = tid >> 5;
    const int lane = tid & 31;
    const int r    = lane & 15;
    const int h    = lane >> 4;
    const int wm   = w >> 2;        // 0..1  -> 64 rows
    const int wn   = w & 3;         // 0..3  -> 32 cols
    const int m0   = blockIdx.y * 128;
    const int n0   = blockIdx.x * 128;

    v8f acc[4][2] = {};

    auto issue = [&](int k0, int buf) {
#pragma unroll
        for (int u = 0; u < 2; ++u) {
            int c = tid * 2 + u;              // 512 chunks of 8 halves (16B)
            int row = c >> 2, col = (c & 3) * 8;
            async_copy16(&As[buf][row][col], A  + (size_t)(m0 + row) * K + k0 + col);
            async_copy16(&Bs[buf][row][col], Bt + (size_t)(n0 + row) * K + k0 + col);
        }
    };

    issue(0, 0);
    async_wait0();
    __syncthreads();

    int cur = 0;
    for (int k0 = 0; k0 < K; k0 += 32) {
        if (k0 + 32 < K) issue(k0 + 32, cur ^ 1);   // prefetch next tile

        v16h am[4], bn[2];
#pragma unroll
        for (int i = 0; i < 4; ++i)
            am[i] = ld_afrag(&As[cur][wm * 64 + i * 16][0], 40, lane);
#pragma unroll
        for (int j = 0; j < 2; ++j)
            bn[j] = ld_bfrag(&Bs[cur][wn * 32 + j * 16][0], 40, lane);
#pragma unroll
        for (int i = 0; i < 4; ++i)
#pragma unroll
            for (int j = 0; j < 2; ++j)
                acc[i][j] = WMMA_F16(am[i], bn[j], acc[i][j]);

        async_wait0();
        __syncthreads();
        cur ^= 1;
    }

    // ---- store: C layout row = 8h + jj, col = r ----
#pragma unroll
    for (int i = 0; i < 4; ++i)
#pragma unroll
        for (int j = 0; j < 2; ++j)
#pragma unroll
            for (int jj = 0; jj < 8; ++jj) {
                int m = m0 + wm * 64 + i * 16 + 8 * h + jj;
                int n = n0 + wn * 32 + j * 16 + r;
                float v = acc[i][j][jj];
                if (OMODE == 0) {
                    int b = m >> 10, s = m & 1023, hh = n >> 6, d = n & 63;
                    ((half_t*)Op)[((((size_t)b * 16 + hh) * 1024 + s) * 64) + d] = (half_t)v;
                } else {
                    ((float*)Op)[(size_t)m * N + n] = v;
                }
            }
}

// ---------------------------------------------------------------------------
// Flash-attention: grid (S/64, H, B), 128 threads = 4 waves (16 q-rows each).
// Q/K tiles stream in via async global->LDS; scores & PV via WMMA;
// online softmax carried in the C-fragment layout.
// ---------------------------------------------------------------------------
__global__ __launch_bounds__(128) void attn_kernel(const half_t* __restrict__ Qh,
                                                   const half_t* __restrict__ Kh,
                                                   const half_t* __restrict__ Vh,
                                                   const float* __restrict__ mat,
                                                   const unsigned char* __restrict__ msk,
                                                   half_t* __restrict__ ctx) {
    __shared__ half_t Qs[64][72];
    __shared__ half_t Ks[32][72];
    __shared__ half_t Vt[64][40];       // transposed: [d][k]
    __shared__ half_t Pw[4][16][40];    // per-wave P bounce buffer

    const int tid  = threadIdx.x;
    const int w    = tid >> 5;
    const int lane = tid & 31;
    const int r    = lane & 15;
    const int h    = lane >> 4;
    const int q0   = blockIdx.x * 64;
    const int head = blockIdx.y;
    const int b    = blockIdx.z;

    const half_t* Qp = Qh + (((size_t)b * 16 + head) * 1024 + q0) * 64;
    const half_t* Kp = Kh + (((size_t)b * 16 + head) * 1024) * 64;
    const half_t* Vp = Vh + (((size_t)b * 16 + head) * 1024) * 64;
    const float* mbase = mat + (size_t)b * 1024 * 1024;
    const unsigned char* kbase = msk + (size_t)b * 1024 * 1024;

    // async-load Q tile 64x64 (f16) once
#pragma unroll
    for (int u = 0; u < 4; ++u) {
        int c = tid * 4 + u;                 // 512 chunks of 8 halves
        int row = c >> 3, col = (c & 7) * 8;
        async_copy16(&Qs[row][col], Qp + (size_t)row * 64 + col);
    }
    async_wait0();
    __syncthreads();
    v16h aQ0 = ld_afrag(&Qs[w * 16][0], 72, lane);
    v16h aQ1 = ld_afrag(&Qs[w * 16][32], 72, lane);

    float mrun[8], lrun[8];
#pragma unroll
    for (int j = 0; j < 8; ++j) { mrun[j] = -3.0e38f; lrun[j] = 0.0f; }
    v8f Oacc[4] = {};

    for (int kc = 0; kc < 1024; kc += 32) {
        // K chunk 32x64 -> LDS [key][d] (pure copy -> async path)
#pragma unroll
        for (int u = 0; u < 2; ++u) {
            int c = tid * 2 + u;             // 256 chunks of 8 halves
            int row = c >> 3, col = (c & 7) * 8;
            async_copy16(&Ks[row][col], Kp + (size_t)(kc + row) * 64 + col);
        }
        // V chunk 32x64 -> LDS transposed [d][key] (needs VGPR bounce)
#pragma unroll
        for (int u = 0; u < 2; ++u) {
            int c = tid * 2 + u;
            int row = c >> 3, col = (c & 7) * 8;
            v8h x = *(const v8h*)(Vp + (size_t)(kc + row) * 64 + col);
#pragma unroll
            for (int i = 0; i < 8; ++i) Vt[col + i][row] = x[i];
        }
        // prefetch next chunk's bias/mask rows into GL2
        if (kc + 32 < 1024) {
            int qp = q0 + w * 16 + 8 * h;
            __builtin_prefetch(mbase + (size_t)qp * 1024 + kc + 32, 0, 1);
            __builtin_prefetch(kbase + (size_t)qp * 1024 + kc + 32, 0, 1);
        }
        async_wait0();
        __syncthreads();

        // scores: [16 q][32 keys] in two 16x16 accumulators
        v8f s0 = {}, s1 = {};
        s0 = WMMA_F16(aQ0, ld_bfrag(&Ks[0][0], 72, lane), s0);
        s0 = WMMA_F16(aQ1, ld_bfrag(&Ks[0][32], 72, lane), s0);
        s1 = WMMA_F16(aQ0, ld_bfrag(&Ks[16][0], 72, lane), s1);
        s1 = WMMA_F16(aQ1, ld_bfrag(&Ks[16][32], 72, lane), s1);

        float corr[8];
#pragma unroll
        for (int j = 0; j < 8; ++j) {
            int q = q0 + w * 16 + 8 * h + j;
            const float* mr = mbase + (size_t)q * 1024 + kc;
            const unsigned char* kr = kbase + (size_t)q * 1024 + kc;
            float x0 = s0[j] * 0.125f * mr[r];
            if (kr[r]) x0 = -1.0e9f;
            float x1 = s1[j] * 0.125f * mr[16 + r];
            if (kr[16 + r]) x1 = -1.0e9f;

            float mx = fmaxf(x0, x1);
#pragma unroll
            for (int off = 1; off < 16; off <<= 1)
                mx = fmaxf(mx, __shfl_xor(mx, off, 32));
            float mnew = fmaxf(mrun[j], mx);
            float c = __expf(mrun[j] - mnew);
            float p0 = __expf(x0 - mnew);
            float p1 = __expf(x1 - mnew);
            float sum = p0 + p1;
#pragma unroll
            for (int off = 1; off < 16; off <<= 1)
                sum += __shfl_xor(sum, off, 32);
            lrun[j] = lrun[j] * c + sum;
            mrun[j] = mnew;
            corr[j] = c;
            Pw[w][8 * h + j][r]      = (half_t)p0;
            Pw[w][8 * h + j][16 + r] = (half_t)p1;
        }
#pragma unroll
        for (int t = 0; t < 4; ++t)
#pragma unroll
            for (int j = 0; j < 8; ++j) Oacc[t][j] *= corr[j];

        // wave-private LDS bounce: C-layout P -> A-fragment layout
        asm volatile("s_wait_dscnt 0" ::: "memory");
        v16h ap = ld_afrag(&Pw[w][0][0], 40, lane);
#pragma unroll
        for (int t = 0; t < 4; ++t)
            Oacc[t] = WMMA_F16(ap, ld_bfrag(&Vt[t * 16][0], 40, lane), Oacc[t]);
        __syncthreads();
    }

    // normalize + store context in [b, s, h*64 + d] (f16, feeds FC GEMM)
#pragma unroll
    for (int t = 0; t < 4; ++t)
#pragma unroll
        for (int j = 0; j < 8; ++j) {
            float v = Oacc[t][j] / lrun[j];
            int q = q0 + w * 16 + 8 * h + j;
            int col = head * 64 + t * 16 + r;
            ctx[((size_t)b * 1024 + q) * 1024 + col] = (half_t)v;
        }
}

// ---------------------------------------------------------------------------
// In-place row LayerNorm over d_out: one block per row of 1024 floats
// ---------------------------------------------------------------------------
__global__ __launch_bounds__(256) void ln_kernel(float* __restrict__ out,
                                                 const float* __restrict__ gamma,
                                                 const float* __restrict__ beta) {
    __shared__ float rs[256], rs2[256];
    const int tid = threadIdx.x;
    float* x = out + (size_t)blockIdx.x * 1024;
    float s = 0.0f, s2 = 0.0f;
#pragma unroll
    for (int i = tid; i < 1024; i += 256) {
        float v = x[i];
        s += v; s2 += v * v;
    }
    rs[tid] = s; rs2[tid] = s2;
    __syncthreads();
    for (int st = 128; st > 0; st >>= 1) {
        if (tid < st) { rs[tid] += rs[tid + st]; rs2[tid] += rs2[tid + st]; }
        __syncthreads();
    }
    float mu  = rs[0] * (1.0f / 1024.0f);
    float var = rs2[0] * (1.0f / 1024.0f) - mu * mu;
    float inv = rsqrtf(var + 1e-5f);
#pragma unroll
    for (int i = tid; i < 1024; i += 256)
        x[i] = (x[i] - mu) * inv * gamma[i] + beta[i];
}

// ---------------------------------------------------------------------------
extern "C" void kernel_launch(void* const* d_in, const int* in_sizes, int n_in,
                              void* d_out, int out_size, void* d_ws, size_t ws_size,
                              hipStream_t stream) {
    (void)in_sizes; (void)n_in; (void)out_size; (void)ws_size;
    const float* inQ = (const float*)d_in[0];
    const float* inK = (const float*)d_in[1];
    const float* inV = (const float*)d_in[2];
    const unsigned char* msk = (const unsigned char*)d_in[3];
    const float* mat = (const float*)d_in[4];
    const float* Wq  = (const float*)d_in[5];
    const float* Wk  = (const float*)d_in[6];
    const float* Wv  = (const float*)d_in[7];
    const float* Wfc = (const float*)d_in[8];
    const float* gamma = (const float*)d_in[9];
    const float* beta  = (const float*)d_in[10];
    float* out = (float*)d_out;

    half_t* ws = (half_t*)d_ws;
    const size_t PROJ = (size_t)8 * 1024 * 1024;      // 8M halves per slot
    half_t* Xq  = ws + 0 * PROJ;
    half_t* Xk  = ws + 1 * PROJ;
    half_t* Xv  = ws + 2 * PROJ;
    half_t* Qh  = ws + 3 * PROJ;
    half_t* Kh  = ws + 4 * PROJ;
    half_t* Vh  = ws + 5 * PROJ;
    half_t* ctx = Xq;                                  // reuse after projections
    half_t* Wqt  = ws + 6 * PROJ;
    half_t* Wkt  = Wqt + (size_t)1024 * 1024;
    half_t* Wvt  = Wkt + (size_t)1024 * 1024;
    half_t* Wfct = Wvt + (size_t)1024 * 1024;

    // one-time precision passes
    cvt_f16<<<8192, 256, 0, stream>>>(inQ, Xq);
    cvt_f16<<<8192, 256, 0, stream>>>(inK, Xk);
    cvt_f16<<<8192, 256, 0, stream>>>(inV, Xv);
    dim3 tgrid(32, 32);
    tcvt_f16<<<tgrid, 256, 0, stream>>>(Wq,  Wqt);
    tcvt_f16<<<tgrid, 256, 0, stream>>>(Wk,  Wkt);
    tcvt_f16<<<tgrid, 256, 0, stream>>>(Wv,  Wvt);
    tcvt_f16<<<tgrid, 256, 0, stream>>>(Wfc, Wfct);

    dim3 gblk(256), ggrid(1024 / 128, 8192 / 128);    // N tiles x M tiles
    gemm_wmma<0><<<ggrid, gblk, 0, stream>>>(Xq, Wqt, Qh, 8192, 1024, 1024);
    gemm_wmma<0><<<ggrid, gblk, 0, stream>>>(Xk, Wkt, Kh, 8192, 1024, 1024);
    gemm_wmma<0><<<ggrid, gblk, 0, stream>>>(Xv, Wvt, Vh, 8192, 1024, 1024);

    attn_kernel<<<dim3(1024 / 64, 16, 8), 128, 0, stream>>>(Qh, Kh, Vh, mat, msk, ctx);

    gemm_wmma<1><<<ggrid, gblk, 0, stream>>>(ctx, Wfct, out, 8192, 1024, 1024);

    ln_kernel<<<8192, 256, 0, stream>>>(out, gamma, beta);
}
